// AGCRNCell_3736621548018
// MI455X (gfx1250) — matrix operand order repaired and checked by reading, not compiled, not run on hardware
//
#include <hip/hip_runtime.h>
#include <hip/hip_bf16.h>
#include <math.h>

// ---------------- problem constants ----------------
#define BATCH   32
#define NNODE   3000
#define NP      3008          // padded to multiple of 64
#define CIN     2
#define HID     64
#define CAT     66            // CIN + HID
#define DEMB    16
#define C2      2112          // BATCH * CAT  (multiple of 64)
#define KCON    198           // 3 * CAT (contraction K)
#define KPAD    224           // 7 * 32

typedef __attribute__((ext_vector_type(16))) __bf16 v16bf;
typedef __attribute__((ext_vector_type(8)))  __bf16 v8bf;
typedef __attribute__((ext_vector_type(8)))  float  v8f;

// ---------------- WMMA helpers (CDNA5 gfx1250, wave32) ----------------
__device__ inline v8f wmma_bf16(v16bf a, v16bf b, v8f c) {
  // D = A(16x32 bf16) * B(32x16 bf16) + C(16x16 f32)
  return __builtin_amdgcn_wmma_f32_16x16x32_bf16(
      /*neg_a=*/false, a, /*neg_b=*/false, b,
      /*c_mod=*/(short)0, c, /*reuse_a=*/false, /*reuse_b=*/false);
}

// Join two contiguous 8-half runs into one 16-half fragment register set.
__device__ inline v16bf ld2(const __bf16* p0, const __bf16* p1) {
  v8bf lo = *(const v8bf*)p0;
  v8bf hi = *(const v8bf*)p1;
  v16bf r;
#pragma unroll
  for (int i = 0; i < 8; ++i) { r[i] = lo[i]; r[i + 8] = hi[i]; }
  return r;
}

// A fragment 16x32 (MxK) from row-major LDS (stride in halves).
// ISA 7.12.2: lane m=lane&15, half=lane>>4; K runs {8h..8h+7},{16+8h..16+8h+7}.
__device__ inline v16bf load_frag_a(const __bf16* base, int stride, int lane) {
  const int m = lane & 15, h = lane >> 4;
  const __bf16* row = base + m * stride;
  return ld2(row + 8 * h, row + 16 + 8 * h);
}

// B fragment 32x16 (KxN) from LDS holding B^T rows (per-column K contiguous).
// lane n=lane&15, half h: K runs {8h..8h+7},{16+8h..16+8h+7} at fixed column n.
__device__ inline v16bf load_frag_b(const __bf16* base, int stride, int lane) {
  const int n = lane & 15, h = lane >> 4;
  const __bf16* row = base + n * stride;
  return ld2(row + 8 * h, row + 16 + 8 * h);
}

// ---------------- kernel 1: S = softmax(relu(ne @ ne^T)) in bf16 -------------
__device__ inline float blk_reduce_max(float v, float* red) {
  int t = threadIdx.x;
  red[t] = v; __syncthreads();
  for (int s = 128; s > 0; s >>= 1) {
    if (t < s) red[t] = fmaxf(red[t], red[t + s]);
    __syncthreads();
  }
  float r = red[0]; __syncthreads();
  return r;
}
__device__ inline float blk_reduce_sum(float v, float* red) {
  int t = threadIdx.x;
  red[t] = v; __syncthreads();
  for (int s = 128; s > 0; s >>= 1) {
    if (t < s) red[t] += red[t + s];
    __syncthreads();
  }
  float r = red[0]; __syncthreads();
  return r;
}

__global__ __launch_bounds__(256)
void support_kernel(const float* __restrict__ ne, __bf16* __restrict__ S) {
  const int n = blockIdx.x;
  const int t = threadIdx.x;
  __shared__ float red[256];
  __shared__ float sne[DEMB];

  if (n >= NNODE) {  // pad rows -> zero
    for (int m = t; m < NP; m += 256) S[(size_t)n * NP + m] = (__bf16)0.0f;
    return;
  }
  if (t < DEMB) sne[t] = ne[n * DEMB + t];
  __syncthreads();

  float mx = 0.0f;
  for (int m = t; m < NNODE; m += 256) {
    const float4* p = (const float4*)(ne + m * DEMB);
    float d = 0.0f;
#pragma unroll
    for (int q = 0; q < 4; ++q) {
      float4 v = p[q];
      d += v.x * sne[4 * q + 0] + v.y * sne[4 * q + 1] +
           v.z * sne[4 * q + 2] + v.w * sne[4 * q + 3];
    }
    mx = fmaxf(mx, fmaxf(d, 0.0f));
  }
  const float rowmax = blk_reduce_max(mx, red);

  float sm = 0.0f;
  for (int m = t; m < NNODE; m += 256) {
    const float4* p = (const float4*)(ne + m * DEMB);
    float d = 0.0f;
#pragma unroll
    for (int q = 0; q < 4; ++q) {
      float4 v = p[q];
      d += v.x * sne[4 * q + 0] + v.y * sne[4 * q + 1] +
           v.z * sne[4 * q + 2] + v.w * sne[4 * q + 3];
    }
    sm += __expf(fmaxf(d, 0.0f) - rowmax);
  }
  const float inv = 1.0f / blk_reduce_sum(sm, red);

  for (int m = t; m < NP; m += 256) {
    float o = 0.0f;
    if (m < NNODE) {
      const float4* p = (const float4*)(ne + m * DEMB);
      float d = 0.0f;
#pragma unroll
      for (int q = 0; q < 4; ++q) {
        float4 v = p[q];
        d += v.x * sne[4 * q + 0] + v.y * sne[4 * q + 1] +
             v.z * sne[4 * q + 2] + v.w * sne[4 * q + 3];
      }
      o = __expf(fmaxf(d, 0.0f) - rowmax) * inv;
    }
    S[(size_t)n * NP + m] = (__bf16)o;
  }
}

// ---------------- kernel 2: XT[b*CAT+c][m] = concat(x,state), bf16 -----------
// Transposed layout: C2 rows x NP cols, node index contiguous.
__global__ __launch_bounds__(256)
void pack_x_kernel(const float* __restrict__ x, const float* __restrict__ state,
                   __bf16* __restrict__ XT) {
  int idx = blockIdx.x * 256 + threadIdx.x;
  if (idx >= C2 * NP) return;
  int j = idx / NP, m = idx % NP;
  int b = j / CAT, c = j % CAT;
  float v = 0.0f;
  if (m < NNODE)
    v = (c < CIN) ? x[(b * NNODE + m) * CIN + c]
                  : state[(b * NNODE + m) * HID + (c - CIN)];
  XT[idx] = (__bf16)v;
}

// ---------------- kernel 3: Y^T = (S @ X)^T via bf16 WMMA --------------------
// A = S (NPxNP row-major). BT = X^T (C2 x NP, K contiguous per column).
// mode 0: Y = S@X ; mode 1: Y = 2*(S@X) - Xsub  (Chebyshev recurrence)
__global__ __launch_bounds__(256)
void gemm_sx_kernel(const __bf16* __restrict__ A, const __bf16* __restrict__ BT,
                    const __bf16* __restrict__ XsubT, __bf16* __restrict__ YT,
                    int mode) {
  __shared__ __bf16 As[64][72];   // 64 S-rows   x 64 K (stride padded)
  __shared__ __bf16 Bs[64][72];   // 64 out-cols x 64 K

  const int t = threadIdx.x;
  const int wave = t >> 5, lane = t & 31;
  const int mi = wave >> 1;            // M sub-tile 0..3
  const int ni0 = (wave & 1) * 2;      // N sub-tiles ni0, ni0+1
  const int rowBase = blockIdx.y * 64;
  const int colBase = blockIdx.x * 64;

  v8f acc0 = {}; v8f acc1 = {};

  const int r = t >> 2, c16 = (t & 3) * 16;     // cooperative tile fill
  const size_t aRow = (size_t)(rowBase + r) * NP;
  const size_t bRow = (size_t)(colBase + r) * NP;

  for (int k0 = 0; k0 < NP; k0 += 64) {
    *(uint4*)&As[r][c16]     = *(const uint4*)&A[aRow + k0 + c16];
    *(uint4*)&As[r][c16 + 8] = *(const uint4*)&A[aRow + k0 + c16 + 8];
    *(uint4*)&Bs[r][c16]     = *(const uint4*)&BT[bRow + k0 + c16];
    *(uint4*)&Bs[r][c16 + 8] = *(const uint4*)&BT[bRow + k0 + c16 + 8];
    if (k0 + 64 < NP) {  // prefetch next K-tile (global_prefetch_b8)
      __builtin_prefetch(&A[aRow + k0 + 64 + c16], 0, 3);
      __builtin_prefetch(&BT[bRow + k0 + 64 + c16], 0, 3);
    }
    __syncthreads();
#pragma unroll
    for (int ks = 0; ks < 2; ++ks) {
      const int ko = ks * 32;
      v16bf af = load_frag_a(&As[mi * 16][ko], 72, lane);
      v16bf b0 = load_frag_b(&Bs[ni0 * 16][ko], 72, lane);
      v16bf b1 = load_frag_b(&Bs[(ni0 + 1) * 16][ko], 72, lane);
      acc0 = wmma_bf16(af, b0, acc0);
      acc1 = wmma_bf16(af, b1, acc1);
    }
    __syncthreads();
  }

  // epilogue: lane owns rows gr0..gr0+7 (contiguous in Y^T) at one column
  const int half = lane >> 4, nn = lane & 15;
  const int gr0 = rowBase + mi * 16 + 8 * half;
  {
    const int gc = colBase + ni0 * 16 + nn;
    v8bf o;
    if (mode) {
      v8bf xs = *(const v8bf*)&XsubT[(size_t)gc * NP + gr0];
#pragma unroll
      for (int j = 0; j < 8; ++j) o[j] = (__bf16)(2.0f * acc0[j] - (float)xs[j]);
    } else {
#pragma unroll
      for (int j = 0; j < 8; ++j) o[j] = (__bf16)acc0[j];
    }
    *(v8bf*)&YT[(size_t)gc * NP + gr0] = o;
  }
  {
    const int gc = colBase + (ni0 + 1) * 16 + nn;
    v8bf o;
    if (mode) {
      v8bf xs = *(const v8bf*)&XsubT[(size_t)gc * NP + gr0];
#pragma unroll
      for (int j = 0; j < 8; ++j) o[j] = (__bf16)(2.0f * acc1[j] - (float)xs[j]);
    } else {
#pragma unroll
      for (int j = 0; j < 8; ++j) o[j] = (__bf16)acc1[j];
    }
    *(v8bf*)&YT[(size_t)gc * NP + gr0] = o;
  }
}

// ---------------- kernel 4: per-node weighted contraction --------------------
// out[b,n,:] = [X|Y1|Y2](b,n,198) @ W_n(198xO) + bias_n ; W_n = ne_n . Wpool
// PHASE 0 (gate,  O=128): z_r=sigmoid(.); store z; rewrite X^T state-cols = r*state
// PHASE 1 (update,O=64):  hc=tanh(.); out = z*state + (1-z)*hc
template <int O, int PHASE>
__global__ __launch_bounds__(256)
void node_gemm_kernel(const __bf16* __restrict__ XT, const __bf16* __restrict__ Y1T,
                      const __bf16* __restrict__ Y2T, const float* __restrict__ ne,
                      const float* __restrict__ wpool, const float* __restrict__ bpool,
                      const float* __restrict__ state, float* __restrict__ Zbuf,
                      __bf16* XT_out, float* __restrict__ out) {
  const int n = blockIdx.x;
  const int t = threadIdx.x;
  const int wave = t >> 5, lane = t & 31;

  __shared__ float sne[DEMB];
  __shared__ float sbias[O];
  __shared__ __bf16 sWT[O][40];   // W_n chunk, TRANSPOSED: o rows x 32 K
  __shared__ __bf16 sA[32][40];   // 32 batch rows x 32 K

  if (t < DEMB) sne[t] = ne[n * DEMB + t];
  __syncthreads();

  if (t < O) {
    float s = 0.0f;
#pragma unroll
    for (int d = 0; d < DEMB; ++d) s += sne[d] * bpool[d * O + t];
    sbias[t] = s;
  }

  constexpr int TILES = (O == 128) ? 2 : 1;
  const int mi = wave & 1;       // batch tile 0..1  (M = 32)
  const int oi0 = wave >> 1;     // output tile base 0..3
  v8f acc[TILES];
#pragma unroll
  for (int q = 0; q < TILES; ++q) acc[q] = (v8f){};

  for (int kc = 0; kc < KPAD / 32; ++kc) {
    const int i0 = kc * 32;
    // W chunk (transposed store): sWT[o][r] = sum_d ne[d]*wpool[(d*198+i0+r)*O+o]
    for (int e = t; e < 32 * O; e += 256) {
      int r = e / O, o = e % O, i = i0 + r;   // o fastest -> coalesced wpool reads
      float s = 0.0f;
      if (i < KCON) {
#pragma unroll
        for (int d = 0; d < DEMB; ++d)
          s += sne[d] * wpool[(d * KCON + i) * O + o];
      }
      sWT[o][r] = (__bf16)s;
    }
    // A chunk: sA[b][kk] = x_g[b, n, i0+kk]  (from transposed X buffers)
    for (int e = t; e < 32 * 32; e += 256) {
      int b = e >> 5, kk = e & 31, i = i0 + kk;
      __bf16 v = (__bf16)0.0f;
      if (i < KCON) {
        int k = i / CAT, c = i % CAT;
        const __bf16* src = (k == 0) ? XT : ((k == 1) ? Y1T : Y2T);
        v = src[(size_t)(b * CAT + c) * NP + n];
      }
      sA[b][kk] = v;
    }
    __syncthreads();

    v16bf af = load_frag_a(&sA[mi * 16][0], 40, lane);
#pragma unroll
    for (int q = 0; q < TILES; ++q) {
      v16bf bf = load_frag_b(&sWT[(oi0 + q * 4) * 16][0], 40, lane);
      acc[q] = wmma_bf16(af, bf, acc[q]);
    }
    __syncthreads();
  }

  const int half = lane >> 4, nn = lane & 15;
#pragma unroll
  for (int q = 0; q < TILES; ++q) {
    const int oi = oi0 + q * 4;
#pragma unroll
    for (int j = 0; j < 8; ++j) {
      int b = mi * 16 + j + 8 * half;
      int o = oi * 16 + nn;
      float v = acc[q][j] + sbias[o];
      if (PHASE == 0) {
        float s = 1.0f / (1.0f + __expf(-v));          // sigmoid
        if (o < HID) {
          Zbuf[((size_t)b * NNODE + n) * HID + o] = s;  // z
        } else {
          int h = o - HID;                              // r
          float st = state[((size_t)b * NNODE + n) * HID + h];
          XT_out[(size_t)(b * CAT + CIN + h) * NP + n] = (__bf16)(s * st);
        }
      } else {
        float hc = tanhf(v);
        float z  = Zbuf[((size_t)b * NNODE + n) * HID + o];
        float st = state[((size_t)b * NNODE + n) * HID + o];
        out[((size_t)b * NNODE + n) * HID + o] = z * st + (1.0f - z) * hc;
      }
    }
  }
}

// ---------------- host launcher ----------------
extern "C" void kernel_launch(void* const* d_in, const int* in_sizes, int n_in,
                              void* d_out, int out_size, void* d_ws, size_t ws_size,
                              hipStream_t stream) {
  const float* x     = (const float*)d_in[0];
  const float* state = (const float*)d_in[1];
  const float* ne    = (const float*)d_in[2];
  const float* gw    = (const float*)d_in[3];
  const float* gb    = (const float*)d_in[4];
  const float* uw    = (const float*)d_in[5];
  const float* ub    = (const float*)d_in[6];
  float* out = (float*)d_out;

  // workspace carve-up (all offsets 256B-aligned)
  char* ws = (char*)d_ws;
  const size_t szS  = (size_t)NP * NP * sizeof(__bf16);   // 18.1 MB
  const size_t szXT = (size_t)C2 * NP * sizeof(__bf16);   // 12.7 MB each
  __bf16* S    = (__bf16*)(ws);
  __bf16* XT   = (__bf16*)(ws + szS);
  __bf16* Y1T  = (__bf16*)(ws + szS + szXT);
  __bf16* Y2T  = (__bf16*)(ws + szS + 2 * szXT);
  float*  Z    = (float*)(ws + szS + 3 * szXT);           // 24.6 MB

  dim3 gGemm(C2 / 64, NP / 64);  // 33 x 47

  // 1. S = softmax(relu(ne ne^T)), bf16, padded
  support_kernel<<<NP, 256, 0, stream>>>(ne, S);

  // 2. X^T = [x | state] as (B*CAT) x NP, bf16
  pack_x_kernel<<<(C2 * NP) / 256, 256, 0, stream>>>(x, state, XT);

  // 3-4. Chebyshev terms: Y1 = S X ; Y2 = 2 S Y1 - X   (all transposed layout)
  gemm_sx_kernel<<<gGemm, 256, 0, stream>>>(S, XT, nullptr, Y1T, 0);
  gemm_sx_kernel<<<gGemm, 256, 0, stream>>>(S, Y1T, XT, Y2T, 1);

  // 5. gate: z stored, X^T state-columns rewritten in place to r*state
  node_gemm_kernel<128, 0><<<NNODE, 256, 0, stream>>>(
      XT, Y1T, Y2T, ne, gw, gb, state, Z, XT, out);

  // 6-7. Chebyshev terms on candidate
  gemm_sx_kernel<<<gGemm, 256, 0, stream>>>(S, XT, nullptr, Y1T, 0);
  gemm_sx_kernel<<<gGemm, 256, 0, stream>>>(S, Y1T, XT, Y2T, 1);

  // 8. update: hc = tanh(.), out = z*state + (1-z)*hc
  node_gemm_kernel<64, 1><<<NNODE, 256, 0, stream>>>(
      XT, Y1T, Y2T, ne, uw, ub, state, Z, nullptr, out);
}
